// HierarchicalTEN_33114197852602
// MI455X (gfx1250) — matrix operand order repaired and checked by reading, not compiled, not run on hardware
//
#include <hip/hip_runtime.h>
#include <hip/hip_bf16.h>

// ---------------- problem constants (from setup_inputs) ----------------
#define VV 32000
#define DD 512
#define LLAY 4
#define NSC 4
#define KK 16
#define BB 2
#define SSEQ 2048
#define DH 2048                 // 4*D
#define BSROWS (BB * SSEQ)      // 4096

typedef __bf16 bf16_t;
typedef __bf16 v16bf __attribute__((ext_vector_type(16)));
typedef __bf16 v8bf  __attribute__((ext_vector_type(8)));
typedef float  v8f   __attribute__((ext_vector_type(8)));
typedef int    v4i_t __attribute__((ext_vector_type(4)));

#define DIV_UP(a, b) (((a) + (b) - 1) / (b))

// Blocked GEMM geometry
#define BM 128
#define BN 64
#define BK 32
#define LDB_PAD 40   // LDS row stride in elements (32 data + 8 pad -> 80B, bank friendly)

#if defined(__has_builtin)
#if __has_builtin(__builtin_amdgcn_global_load_async_to_lds_b128)
#define HAVE_ASYNC_LDS 1
#else
#define HAVE_ASYNC_LDS 0
#endif
#if __has_builtin(__builtin_amdgcn_s_wait_asynccnt)
#define HAVE_WAIT_ASYNC 1
#else
#define HAVE_WAIT_ASYNC 0
#endif
#else
#define HAVE_ASYNC_LDS 0
#define HAVE_WAIT_ASYNC 0
#endif

// async-to-LDS builtin takes: (AS1 int4* global src, AS3 int4* lds dst, imm offset, imm cpol)
typedef __attribute__((address_space(1))) v4i_t* as1_p;
typedef __attribute__((address_space(3))) v4i_t* as3_p;

template <int N>
__device__ __forceinline__ void wait_asynccnt() {
#if HAVE_WAIT_ASYNC
  __builtin_amdgcn_s_wait_asynccnt(N);
#elif HAVE_ASYNC_LDS
  asm volatile("s_wait_asynccnt %0" ::"i"(N) : "memory");
#endif
}

// ---------------- fragment loads (WMMA 16x16x32 bf16, contiguous k = 16*(lane/16)+s in A and B) ----------------
// A (16x32, row-major source, lda in elements):
//   lane l: row = l&15 ; lanes<16 hold K k0..k0+15 ; lanes>=16 hold K k0+16..k0+31
__device__ __forceinline__ v16bf load_a_frag(const bf16_t* __restrict__ A, int lda,
                                             int row0, int k0, int lane) {
  const bf16_t* p = A + (size_t)(row0 + (lane & 15)) * lda + k0 + ((lane & 16) ? 16 : 0);
  v8bf lo = *reinterpret_cast<const v8bf*>(p);
  v8bf hi = *reinterpret_cast<const v8bf*>(p + 8);
  v16bf f;
#pragma unroll
  for (int i = 0; i < 8; ++i) { f[i] = lo[i]; f[i + 8] = hi[i]; }
  return f;
}
// B = W^T where W is (N,K) row-major. lane l: col n = n0+(l&15);
//   lanes<16 hold K k0..k0+15 ; lanes>=16 hold K k0+16..k0+31
__device__ __forceinline__ v16bf load_b_frag(const bf16_t* __restrict__ W, int ldw,
                                             int n0, int k0, int lane) {
  const bf16_t* p = W + (size_t)(n0 + (lane & 15)) * ldw + k0 + ((lane & 16) ? 16 : 0);
  v8bf lo = *reinterpret_cast<const v8bf*>(p);
  v8bf hi = *reinterpret_cast<const v8bf*>(p + 8);
  v16bf f;
#pragma unroll
  for (int i = 0; i < 8; ++i) { f[i] = lo[i]; f[i + 8] = hi[i]; }
  return f;
}
// B fragment from LDS-staged panel: panel row = local col (0..63), LDB_PAD stride.
__device__ __forceinline__ v16bf load_b_lds(const bf16_t* __restrict__ bp,
                                            int nt, int lane) {
  const bf16_t* p = bp + (size_t)((nt << 4) + (lane & 15)) * LDB_PAD +
                    ((lane & 16) ? 16 : 0);
  v8bf lo = *reinterpret_cast<const v8bf*>(p);
  v8bf hi = *reinterpret_cast<const v8bf*>(p + 8);
  v16bf f;
#pragma unroll
  for (int i = 0; i < 8; ++i) { f[i] = lo[i]; f[i + 8] = hi[i]; }
  return f;
}

__device__ __forceinline__ float gelu_erf(float v) {
  return 0.5f * v * (1.0f + erff(v * 0.70710678118654752f));
}

// ---------------- simple WMMA GEMM (one wave per 16x16 tile) for small N ----------------
__global__ __launch_bounds__(256) void k_gemm_wmma(
    const bf16_t* __restrict__ A, int lda,
    const bf16_t* __restrict__ W, int ldw,
    const float* __restrict__ bias,
    float* __restrict__ Cf, int ldc,
    int M, int N, int K, int act) {
  int wave = blockIdx.x * (blockDim.x >> 5) + (threadIdx.x >> 5);
  int lane = threadIdx.x & 31;
  int ntn = N >> 4;
  int total = (M >> 4) * ntn;
  if (wave >= total) return;
  int tm = wave / ntn, tn = wave - tm * ntn;
  int row0 = tm << 4, n0 = tn << 4;

  v8f acc = {};
  for (int k0 = 0; k0 < K; k0 += 32) {
    v16bf a = load_a_frag(A, lda, row0, k0, lane);
    v16bf b = load_b_frag(W, ldw, n0, k0, lane);
    acc = __builtin_amdgcn_wmma_f32_16x16x32_bf16(false, a, false, b,
                                                  (short)0, acc, false, false);
  }
  int col = n0 + (lane & 15);
  int rbase = row0 + ((lane & 16) ? 8 : 0);
  float bv = bias ? bias[col] : 0.0f;
#pragma unroll
  for (int j = 0; j < 8; ++j) {
    float v = acc[j] + bv;
    if (act == 1) v = gelu_erf(v);
    Cf[(size_t)(rbase + j) * ldc + col] = v;
  }
}

// ---------------- blocked WMMA GEMM with LDS-staged, double-buffered B panel ----------
// Block: 256 threads (8 waves). Output tile BM x BN = 128 x 64.
// Wave w handles rows [w*16, w*16+16), all 64 cols -> 4 accumulators, A reused 4x.
// B panel (64 x 32 bf16) staged to LDS per K-step via async-to-LDS when available.
__global__ __launch_bounds__(256) void k_gemm_lds(
    const bf16_t* __restrict__ A, int lda,
    const bf16_t* __restrict__ W, int ldw,
    const float* __restrict__ bias,
    float* __restrict__ Cf, int ldc,
    bf16_t* __restrict__ Cb, int ldcb,
    int M, int N, int K, int act) {
  __shared__ bf16_t bsm[2][64 * LDB_PAD];
  int tid = threadIdx.x;
  int wid = tid >> 5, lane = tid & 31;
  int nbn = N / BN;
  int bm = blockIdx.x / nbn, bn = blockIdx.x - bm * nbn;
  int row0 = bm * BM + (wid << 4);
  int n0 = bn * BN;

  // cooperative B fill: 256 threads x 16B = 4KB = 64 rows x 32 bf16
  int frow = tid >> 2;            // 0..63 : W row n0+frow
  int fchk = (tid & 3) << 3;      // element offset 0/8/16/24 within the 32-wide K chunk
  const bf16_t* gsrc = W + (size_t)(n0 + frow) * ldw + fchk;
  bf16_t* ldst_base[2] = {&bsm[0][frow * LDB_PAD + fchk], &bsm[1][frow * LDB_PAD + fchk]};

  int KT = K / BK;
  // prime buffer 0
#if HAVE_ASYNC_LDS
  __builtin_amdgcn_global_load_async_to_lds_b128(
      (as1_p)(const void*)(gsrc), (as3_p)(void*)(ldst_base[0]), 0, 0);
#else
  {
    v8bf v = *reinterpret_cast<const v8bf*>(gsrc);
    *reinterpret_cast<v8bf*>(ldst_base[0]) = v;
  }
#endif

  v8f acc[4] = {};
  for (int kt = 0; kt < KT; ++kt) {
    int k0 = kt * BK;
    if (kt + 1 < KT) {
#if HAVE_ASYNC_LDS
      __builtin_amdgcn_global_load_async_to_lds_b128(
          (as1_p)(const void*)(gsrc + k0 + BK),
          (as3_p)(void*)(ldst_base[(kt + 1) & 1]), 0, 0);
#else
      v8bf v = *reinterpret_cast<const v8bf*>(gsrc + k0 + BK);
      *reinterpret_cast<v8bf*>(ldst_base[(kt + 1) & 1]) = v;
#endif
      // prefetch next A fragment line into cache
      __builtin_prefetch(A + (size_t)(row0 + (lane & 15)) * lda + k0 + BK, 0, 1);
    }
#if HAVE_ASYNC_LDS
    if (kt + 1 < KT) wait_asynccnt<1>(); else wait_asynccnt<0>();
#endif
    __syncthreads();               // current buffer visible to all waves

    v16bf a = load_a_frag(A, lda, row0, k0, lane);
    const bf16_t* bp = bsm[kt & 1];
#pragma unroll
    for (int nt = 0; nt < 4; ++nt) {
      v16bf b = load_b_lds(bp, nt, lane);
      acc[nt] = __builtin_amdgcn_wmma_f32_16x16x32_bf16(false, a, false, b,
                                                        (short)0, acc[nt], false, false);
    }
    __syncthreads();               // all waves done reading before overwrite
  }

  // epilogue: 4 tiles per wave
  int colb = lane & 15;
  int rbase = row0 + ((lane & 16) ? 8 : 0);
#pragma unroll
  for (int nt = 0; nt < 4; ++nt) {
    int col = n0 + (nt << 4) + colb;
    float bv = bias ? bias[col] : 0.0f;
#pragma unroll
    for (int j = 0; j < 8; ++j) {
      float v = acc[nt][j] + bv;
      if (act == 1) v = gelu_erf(v);
      if (Cf) Cf[(size_t)(rbase + j) * ldc + col] = v;
      if (Cb) Cb[(size_t)(rbase + j) * ldcb + col] = (bf16_t)v;
    }
  }
}

// ---------------- LayerNorm over rows of width D (one wave per row) ----------------
__global__ __launch_bounds__(256) void k_layernorm(
    const float* __restrict__ in, const float* __restrict__ res,
    const float* __restrict__ g, const float* __restrict__ b,
    float* __restrict__ outf, bf16_t* __restrict__ outb, int rows, int D) {
  int row = blockIdx.x * (blockDim.x >> 5) + (threadIdx.x >> 5);
  if (row >= rows) return;
  int lane = threadIdx.x & 31;
  const float* ip = in + (size_t)row * D;
  const float* rp = res ? res + (size_t)row * D : nullptr;
  int per = D >> 5;           // D=512 -> 16 per lane
  float vals[16];
  float sum = 0.0f;
  for (int j = 0; j < per; ++j) {
    float v = ip[lane + (j << 5)];
    if (rp) v += rp[lane + (j << 5)];
    vals[j] = v;
    sum += v;
  }
  for (int o = 16; o > 0; o >>= 1) sum += __shfl_xor(sum, o);
  float mean = sum / (float)D;
  float vs = 0.0f;
  for (int j = 0; j < per; ++j) { float d = vals[j] - mean; vs += d * d; }
  for (int o = 16; o > 0; o >>= 1) vs += __shfl_xor(vs, o);
  float rstd = rsqrtf(vs / (float)D + 1e-5f);
  for (int j = 0; j < per; ++j) {
    int c = lane + (j << 5);
    float y = (vals[j] - mean) * rstd * g[c] + b[c];
    if (outf) outf[(size_t)row * D + c] = y;
    if (outb) outb[(size_t)row * D + c] = (bf16_t)y;
  }
}

// ---------------- embedding gather + positional add ----------------
__global__ void k_embed(const float* __restrict__ emb, const float* __restrict__ pos,
                        const int* __restrict__ tokens, float* __restrict__ xf,
                        bf16_t* __restrict__ xb) {
  size_t idx = (size_t)blockIdx.x * blockDim.x + threadIdx.x;
  size_t total = (size_t)BSROWS * DD;
  if (idx >= total) return;
  int row = (int)(idx / DD), d = (int)(idx % DD);
  int s = row % SSEQ;
  int tok = tokens[row];
  float v = emb[(size_t)tok * DD + d] + pos[(size_t)s * DD + d];
  xf[idx] = v;
  xb[idx] = (bf16_t)v;
}

// ---------------- avg-pool1d (kernel=stride=sc) ----------------
__global__ void k_downsample(const float* __restrict__ xf, bf16_t* __restrict__ xd,
                             int T, int sc) {
  size_t idx = (size_t)blockIdx.x * blockDim.x + threadIdx.x;
  size_t total = (size_t)BB * T * DD;
  if (idx >= total) return;
  int d = (int)(idx % DD);
  int rt = (int)(idx / DD);
  int t = rt % T, b = rt / T;
  const float* p = xf + ((size_t)b * SSEQ + (size_t)t * sc) * DD + d;
  float s = 0.0f;
  for (int j = 0; j < sc; ++j) s += p[(size_t)j * DD];
  xd[idx] = (bf16_t)(s / (float)sc);
}

// ---------------- linear interp (align_corners=False) into concat buffer ----------------
__global__ void k_upsample_cat(const float* __restrict__ o, bf16_t* __restrict__ catb,
                               int T, int si) {
  size_t idx = (size_t)blockIdx.x * blockDim.x + threadIdx.x;
  size_t total = (size_t)BB * SSEQ * DD;
  if (idx >= total) return;
  int d = (int)(idx % DD);
  int rs = (int)(idx / DD);
  int s = rs % SSEQ, b = rs / SSEQ;
  float val;
  if (T == SSEQ) {
    val = o[((size_t)b * SSEQ + s) * DD + d];
  } else {
    float posf = (s + 0.5f) * ((float)T / (float)SSEQ) - 0.5f;
    float lof = floorf(posf);
    float w = posf - lof;
    int lo = min(max((int)lof, 0), T - 1);
    int hi = min(max((int)lof + 1, 0), T - 1);
    val = o[((size_t)b * T + lo) * DD + d] * (1.0f - w) +
          o[((size_t)b * T + hi) * DD + d] * w;
  }
  catb[((size_t)b * SSEQ + s) * DH + (size_t)si * DD + d] = (bf16_t)val;
}

// ---------------- complex eigenstate recurrence (one wave32 per batch) ----------------
// lanes 0-15: real state sr[i]; lanes 16-31: imag state si[i].
__global__ void k_scan(const float* __restrict__ beta, const float* __restrict__ alpha,
                       const float* __restrict__ omega, const float* __restrict__ R,
                       bf16_t* __restrict__ srb, int T) {
  int b = blockIdx.x;
  int lane = threadIdx.x;        // 32 threads
  int i = lane & 15;
  bool hi = lane >= 16;
  float mag = 1.0f / (1.0f + expf(-alpha[i]));
  float cp = cosf(omega[i]), sp = sinf(omega[i]);
  float Rcol[16];
#pragma unroll
  for (int r = 0; r < 16; ++r) Rcol[r] = R[r * 16 + i];   // column i of R
  float st = 0.0f;
  const float* bp = beta + (size_t)b * T * KK;
  bf16_t* op = srb + (size_t)b * T * 32;
  for (int t = 0; t < T; ++t) {
    float other = __shfl_xor(st, 16);   // imag for low half, real for high half
    float nv;
    if (!hi) nv = mag * (st * cp - other * sp) + bp[(size_t)t * KK + i];
    else     nv = mag * (other * sp + st * cp);
    float acc = 0.0f;
#pragma unroll
    for (int r = 0; r < 16; ++r) acc += __shfl(nv, (lane & 16) + r) * Rcol[r];
    st = acc;
    if (!hi) op[(size_t)t * 32 + i] = (bf16_t)st;   // cols 16..31 stay zero (K pad)
  }
}

// ---------------- dtype conversion helpers ----------------
__global__ void k_convert(const float* __restrict__ s, bf16_t* __restrict__ d, size_t n) {
  size_t i = (size_t)blockIdx.x * blockDim.x + threadIdx.x;
  if (i < n) d[i] = (bf16_t)s[i];
}
__global__ void k_convert_pad(const float* __restrict__ s, bf16_t* __restrict__ d,
                              int rows, int cols, int ldd) {
  int i = blockIdx.x * blockDim.x + threadIdx.x;
  if (i >= rows * ldd) return;
  int r = i / ldd, c = i - r * ldd;
  d[i] = (c < cols) ? (bf16_t)s[r * cols + c] : (bf16_t)0.0f;
}
__global__ void k_zero_bf16(bf16_t* __restrict__ d, size_t n) {
  size_t i = (size_t)blockIdx.x * blockDim.x + threadIdx.x;
  if (i < n) d[i] = (bf16_t)0.0f;
}

// ---------------- host-side launch helpers ----------------
static void gemm(const bf16_t* A, int lda, const bf16_t* W, int ldw,
                 const float* bias, float* Cf, int ldc, bf16_t* Cb, int ldcb,
                 int M, int N, int K, int act, hipStream_t stream) {
  if ((M % BM) == 0 && (N % BN) == 0 && (K % BK) == 0) {
    int blocks = (M / BM) * (N / BN);
    k_gemm_lds<<<blocks, 256, 0, stream>>>(A, lda, W, ldw, bias, Cf, ldc, Cb, ldcb,
                                           M, N, K, act);
  } else {
    int tiles = (M >> 4) * (N >> 4);
    k_gemm_wmma<<<DIV_UP(tiles, 8), 256, 0, stream>>>(A, lda, W, ldw, bias,
                                                      Cf, ldc, M, N, K, act);
  }
}
static void layernorm(const float* in, const float* res, const float* g, const float* b,
                      float* outf, bf16_t* outb, int rows, hipStream_t stream) {
  k_layernorm<<<DIV_UP(rows, 8), 256, 0, stream>>>(in, res, g, b, outf, outb, rows, DD);
}
static void convert(const float* s, bf16_t* d, size_t n, hipStream_t stream) {
  k_convert<<<(int)DIV_UP(n, (size_t)256), 256, 0, stream>>>(s, d, n);
}

extern "C" void kernel_launch(void* const* d_in, const int* in_sizes, int n_in,
                              void* d_out, int out_size, void* d_ws, size_t ws_size,
                              hipStream_t stream) {
  const float* emb   = (const float*)d_in[0];
  const float* pos   = (const float*)d_in[1];
  const float* alpha = (const float*)d_in[2];
  const float* omega = (const float*)d_in[3];
  const float* inW   = (const float*)d_in[4];
  const float* resR  = (const float*)d_in[5];
  const float* outW  = (const float*)d_in[6];
  const float* outb  = (const float*)d_in[7];
  const float* cg    = (const float*)d_in[8];
  const float* cb    = (const float*)d_in[9];
  const float* mixW  = (const float*)d_in[10];
  const float* mixb  = (const float*)d_in[11];
  const float* f1W   = (const float*)d_in[12];
  const float* f1b   = (const float*)d_in[13];
  const float* f2W   = (const float*)d_in[14];
  const float* f2b   = (const float*)d_in[15];
  const float* n1g   = (const float*)d_in[16];
  const float* n1b   = (const float*)d_in[17];
  const float* n2g   = (const float*)d_in[18];
  const float* n2b   = (const float*)d_in[19];
  const int*   toks  = (const int*)d_in[20];
  float* logits = (float*)d_out;

  // ---- carve workspace ----
  char* base = (char*)d_ws;
  size_t off = 0;
  auto carve = [&](size_t bytes) {
    void* p = base + off;
    off += (bytes + 255) & ~(size_t)255;
    return p;
  };
  float*  xf    = (float*)carve((size_t)BSROWS * DD * 4);   // residual stream f32
  bf16_t* xb    = (bf16_t*)carve((size_t)BSROWS * DD * 2);  // bf16 copy of x
  float*  tmpf  = (float*)carve((size_t)BSROWS * DD * 4);   // pre-LN GEMM outputs
  bf16_t* catb  = (bf16_t*)carve((size_t)BSROWS * DH * 2);  // concat of 4 scale outputs
  bf16_t* hb    = (bf16_t*)carve((size_t)BSROWS * DH * 2);  // FFN hidden (bf16)
  bf16_t* xdb   = (bf16_t*)carve((size_t)BSROWS * DD * 2);  // downsampled input
  float*  beta  = (float*)carve((size_t)BSROWS * KK * 4);   // input projections
  bf16_t* srb   = (bf16_t*)carve((size_t)BSROWS * 32 * 2);  // states, K padded 16->32
  float*  of32  = (float*)carve((size_t)BSROWS * DD * 4);   // cell output
  bf16_t* embb  = (bf16_t*)carve((size_t)VV * DD * 2);      // bf16 vocab matrix
  bf16_t* mixWb = (bf16_t*)carve((size_t)DD * DH * 2);
  bf16_t* f1Wb  = (bf16_t*)carve((size_t)DH * DD * 2);
  bf16_t* f2Wb  = (bf16_t*)carve((size_t)DD * DH * 2);
  bf16_t* inWb  = (bf16_t*)carve((size_t)KK * DD * 2);
  bf16_t* outWp = (bf16_t*)carve((size_t)DD * 32 * 2);      // outW padded K 16->32
  (void)ws_size; (void)n_in; (void)in_sizes; (void)out_size;

  const int scales[NSC] = {1, 2, 4, 8};

  // per call: zero K-pad region of the state buffer, convert vocab matrix
  k_zero_bf16<<<DIV_UP(BSROWS * 32, 256), 256, 0, stream>>>(srb, (size_t)BSROWS * 32);
  convert(emb, embb, (size_t)VV * DD, stream);

  // x = emb[tokens] + pos[:, :S]
  k_embed<<<DIV_UP(BSROWS * DD, 256), 256, 0, stream>>>(emb, pos, toks, xf, xb);

  for (int l = 0; l < LLAY; ++l) {
    // ----- multi-scale temporal flow cells -----
    for (int si = 0; si < NSC; ++si) {
      int sc = scales[si];
      int T = SSEQ / sc;
      int rows = BB * T;
      const bf16_t* Ain = xb;
      if (sc > 1) {
        k_downsample<<<DIV_UP(rows * DD, 256), 256, 0, stream>>>(xf, xdb, T, sc);
        Ain = xdb;
      }
      // beta = xd @ inW^T   (M=rows, N=16, K=512) -> simple WMMA path
      convert(inW + (size_t)(l * NSC + si) * KK * DD, inWb, (size_t)KK * DD, stream);
      gemm(Ain, DD, inWb, DD, nullptr, beta, KK, nullptr, 0, rows, KK, DD, 0, stream);
      // sequential complex recurrence -> sr_t (bf16, K padded to 32)
      k_scan<<<BB, 32, 0, stream>>>(beta,
                                    alpha + (size_t)(l * NSC + si) * KK,
                                    omega + (size_t)(l * NSC + si) * KK,
                                    resR + (size_t)(l * NSC + si) * KK * KK,
                                    srb, T);
      // o = LN(sr @ outW^T + outb)   (M=rows, N=512, K=32 padded)
      k_convert_pad<<<DIV_UP(DD * 32, 256), 256, 0, stream>>>(
          outW + (size_t)(l * NSC + si) * DD * KK, outWp, DD, KK, 32);
      gemm(srb, 32, outWp, 32, outb + (size_t)(l * NSC + si) * DD,
           of32, DD, nullptr, 0, rows, DD, 32, 0, stream);
      layernorm(of32, nullptr, cg + (size_t)(l * NSC + si) * DD,
                cb + (size_t)(l * NSC + si) * DD, of32, nullptr, rows, stream);
      // upsample into concat buffer column block si
      k_upsample_cat<<<DIV_UP(BSROWS * DD, 256), 256, 0, stream>>>(of32, catb, T, si);
    }
    // mixed = cat @ mixW^T + mixb ; x = LN(x + mixed)
    convert(mixW + (size_t)l * DD * DH, mixWb, (size_t)DD * DH, stream);
    gemm(catb, DH, mixWb, DH, mixb + (size_t)l * DD,
         tmpf, DD, nullptr, 0, BSROWS, DD, DH, 0, stream);
    layernorm(tmpf, xf, n1g + (size_t)l * DD, n1b + (size_t)l * DD,
              xf, xb, BSROWS, stream);
    // h = gelu(x @ f1W^T + f1b)
    convert(f1W + (size_t)l * DH * DD, f1Wb, (size_t)DH * DD, stream);
    gemm(xb, DD, f1Wb, DD, f1b + (size_t)l * DH,
         nullptr, 0, hb, DH, BSROWS, DH, DD, 1, stream);
    // x = LN(x + h @ f2W^T + f2b)
    convert(f2W + (size_t)l * DD * DH, f2Wb, (size_t)DD * DH, stream);
    gemm(hb, DH, f2Wb, DH, f2b + (size_t)l * DD,
         tmpf, DD, nullptr, 0, BSROWS, DD, DH, 0, stream);
    layernorm(tmpf, xf, n2g + (size_t)l * DD, n2b + (size_t)l * DD,
              xf, xb, BSROWS, stream);
  }

  // logits = x @ emb^T   (M=4096, N=32000, K=512)
  gemm(xb, DD, embb, DD, nullptr, logits, VV, nullptr, 0,
       BSROWS, VV, DD, 0, stream);
}
